// VAFlowNet_19885698581174
// MI455X (gfx1250) — matrix operand order repaired and checked by reading, not compiled, not run on hardware
//
#include <hip/hip_runtime.h>
#include <math.h>

typedef __attribute__((ext_vector_type(16))) __bf16 v16bf;
typedef __attribute__((ext_vector_type(8)))  __bf16 v8bf;
typedef __attribute__((ext_vector_type(4)))  __bf16 v4bf;
typedef __attribute__((ext_vector_type(2)))  __bf16 v2bf;
typedef __attribute__((ext_vector_type(8)))  float  v8f;
typedef __attribute__((ext_vector_type(2)))  float  v2f;

#define HH 16
#define WW 16
#define NPIX 256

// ---------------------------------------------------------------------------
// Implicit-GEMM 3x3 SAME conv. One wave per (n, oc-tile, image row).
// GEMM: M=16 oc, N=16 pixels, K = CIN*9 with k = ic*9 + t (t = dy*3+dx),
// so weights for an oc row are CONTIGUOUS in global memory.
// Pipeline:
//   1) zero LDS
//   2) async-copy 3 padded input rows global->LDS (f32, b128 granularity)
//      + stage oc-tile weights global->LDS as bf16 (vector ds stores)
//   3) LDS transpose pass: build B-panel xk[col][K] in bf16
//   4) fully unrolled K loop: 2x ds_load_b128 (A) + 2x ds_load_b128 (B)
//      + v_wmma_f32_16x16x32_bf16
// ACT: 0 = none, 1 = leaky_relu(0.01)
// ---------------------------------------------------------------------------
template <int CIN, int COUT, int ACT>
__global__ __launch_bounds__(32) void conv3x3_wmma(
    const float* __restrict__ xin, const float* __restrict__ wgt,
    const float* __restrict__ bias, float* __restrict__ yout)
{
  constexpr int KTOT = CIN * 9;
  constexpr int KPAD = (KTOT + 31) & ~31;
  constexpr int CHUNKS = KPAD / 32;
  constexpr int SXW = 24;                 // padded row stride (floats): col c -> idx 4+c

  __shared__ alignas(16) float  sx[3 * CIN * SXW];   // f32 halo rows (async dest)
  __shared__ alignas(16) __bf16 wlds[16 * KPAD];     // A panel: [ocl][k]
  __shared__ alignas(16) __bf16 xk[16 * KPAD];       // B panel: [col][k]

  const int n = blockIdx.x, ot = blockIdx.y, row = blockIdx.z;
  const int lane = threadIdx.x;
  const int m16 = lane & 15, g = lane >> 4;

  // ---- 1) zero all LDS (b128 stores) ----
  {
    float4 z4; z4.x = z4.y = z4.z = z4.w = 0.0f;
    float4* p1 = (float4*)sx;
    for (int i = lane; i < 3 * CIN * SXW / 4; i += 32) p1[i] = z4;
    float4* p2 = (float4*)wlds;
    for (int i = lane; i < 16 * KPAD * 2 / 16; i += 32) p2[i] = z4;
    float4* p3 = (float4*)xk;
    for (int i = lane; i < 16 * KPAD * 2 / 16; i += 32) p3[i] = z4;
  }
  // DS zero-stores must land before async engine writes the same LDS region
  asm volatile("s_wait_dscnt 0x0" ::: "memory");
  __syncthreads();

  // ---- 2a) async-copy interior of the 3 halo rows: global -> LDS ----
  {
    const int nq = 3 * CIN * 4;           // b128 quads (4 floats each)
    for (int idx = lane; idx < nq; idx += 32) {
      const int rr  = idx / (CIN * 4);
      const int rem = idx - rr * (CIN * 4);
      const int ch  = rem >> 2, q = rem & 3;
      const int srow = row + rr - 1;
      if (srow >= 0 && srow < HH) {
        const unsigned lo = (unsigned)(size_t)&sx[(rr * CIN + ch) * SXW + 4 + 4 * q];
        const unsigned long long ga = (unsigned long long)(size_t)
            (xin + (size_t)(n * CIN + ch) * NPIX + srow * WW + 4 * q);
        asm volatile("global_load_async_to_lds_b128 %0, %1, off"
                     :: "v"(lo), "v"(ga) : "memory");
      }
    }
  }

  // ---- 2b) stage oc-tile weights as bf16 (overlaps with async) ----
  {
    const int ocl = lane & 15;
    const int oc = ot * 16 + ocl;
    #pragma unroll 4
    for (int q = (lane >> 4); q < KTOT / 4; q += 2) {
      v4bf d;
      if (oc < COUT) {
        const float4 wq = *(const float4*)&wgt[(size_t)oc * KTOT + 4 * q];
        d[0] = (__bf16)wq.x; d[1] = (__bf16)wq.y;
        d[2] = (__bf16)wq.z; d[3] = (__bf16)wq.w;
      } else {
        d[0] = d[1] = d[2] = d[3] = (__bf16)0.0f;
      }
      *(v4bf*)&wlds[ocl * KPAD + 4 * q] = d;
    }
  }

  asm volatile("s_wait_asynccnt 0x0" ::: "memory");
  __syncthreads();

  // ---- 3) transpose pass: xk[col][k] = x[ic][row+dy-1][col+dx-1], k = ic*9+t ----
  {
    const int col = lane & 15, half = lane >> 4;
    constexpr int KH = KTOT / 2;          // k-range per half-wave
    #pragma unroll 4
    for (int p = 0; p < KH / 2; ++p) {
      const int k0 = half * KH + 2 * p;
      const int ic0 = k0 / 9,       t0 = k0 - ic0 * 9;
      const int ic1 = (k0 + 1) / 9, t1 = (k0 + 1) - ic1 * 9;
      // LDS row idx = 4 + (col + dx - 1) = 3 + col + t%3
      const float f0 = sx[((t0 / 3) * CIN + ic0) * SXW + 3 + col + (t0 % 3)];
      const float f1 = sx[((t1 / 3) * CIN + ic1) * SXW + 3 + col + (t1 % 3)];
      v2bf pr; pr[0] = (__bf16)f0; pr[1] = (__bf16)f1;
      *(v2bf*)&xk[col * KPAD + k0] = pr;
    }
  }
  __syncthreads();

  // ---- 4) MMA loop: fully unrolled, vector LDS loads, no branches ----
  v8f acc = {};
  #pragma unroll
  for (int ck = 0; ck < CHUNKS; ++ck) {
    // A fragment: per-lane k runs [32ck+8g, +8) and [32ck+16+8g, +8)
    const v8bf a0 = *(const v8bf*)&wlds[m16 * KPAD + 32 * ck + 8 * g];
    const v8bf a1 = *(const v8bf*)&wlds[m16 * KPAD + 32 * ck + 16 + 8 * g];
    // B fragment: per-lane k runs [32ck+8g, +8) and [32ck+16+8g, +8)
    const v8bf b0 = *(const v8bf*)&xk[m16 * KPAD + 32 * ck + 8 * g];
    const v8bf b1 = *(const v8bf*)&xk[m16 * KPAD + 32 * ck + 16 + 8 * g];
    const v16bf af = __builtin_shufflevector(a0, a1, 0,1,2,3,4,5,6,7,8,9,10,11,12,13,14,15);
    const v16bf bf = __builtin_shufflevector(b0, b1, 0,1,2,3,4,5,6,7,8,9,10,11,12,13,14,15);
    acc = __builtin_amdgcn_wmma_f32_16x16x32_bf16(false, af, false, bf,
                                                  (short)0, acc, false, false);
  }

  // ---- epilogue: C/D layout M = v + 8g, N = lane%16 ----
  #pragma unroll
  for (int v = 0; v < 8; ++v) {
    const int oc = ot * 16 + v + 8 * g;
    if (oc < COUT) {
      float val = acc[v] + bias[oc];
      if (ACT == 1) val = (val > 0.0f) ? val : 0.01f * val;
      yout[(size_t)(n * COUT + oc) * NPIX + row * WW + m16] = val;
    }
  }
}

// ---------------------------------------------------------------------------
// Squeeze-excite path (tiny): pooled 1x1 convs, sigmoid. se: [n][16]
// ---------------------------------------------------------------------------
__global__ __launch_bounds__(64) void se_kernel(
    const float* __restrict__ x, const float* __restrict__ w1,
    const float* __restrict__ b1, const float* __restrict__ w2,
    const float* __restrict__ b2, float* __restrict__ sev)
{
  const int n = blockIdx.x, tid = threadIdx.x;
  __shared__ float pooled[64], h1[32];
  if (tid < 64) {
    const float* xp = x + (size_t)(n * 64 + tid) * NPIX;
    float s = 0.0f;
    for (int i = 0; i < NPIX; ++i) s += xp[i];
    pooled[tid] = s * (1.0f / 256.0f);
  }
  __syncthreads();
  if (tid < 32) {
    float s = b1[tid];
    for (int c = 0; c < 64; ++c) s += w1[tid * 64 + c] * pooled[c];
    h1[tid] = (s > 0.0f) ? s : 0.01f * s;
  }
  __syncthreads();
  if (tid < 16) {
    float s = b2[tid];
    for (int j = 0; j < 32; ++j) s += w2[tid * 32 + j] * h1[j];
    sev[n * 16 + tid] = 1.0f / (1.0f + __expf(-s));
  }
}

// ---------------------------------------------------------------------------
// ATA init: zeros + 1e-12 jitter on the diagonal; ATB zeroed.
// ---------------------------------------------------------------------------
__global__ void ata_init(float* __restrict__ ATA, float* __restrict__ ATB)
{
  const size_t idx = (size_t)blockIdx.x * blockDim.x + threadIdx.x;
  if (idx < (size_t)16 * 1048576) {
    const size_t w = idx & 1048575u;
    const int r = (int)(w >> 10), c = (int)(w & 1023u);
    ATA[idx] = (r == c) ? 1e-12f : 0.0f;
  }
  if (idx < 16 * 1024) ATB[idx] = 0.0f;
}

// ---------------------------------------------------------------------------
// Sparse assembly of ATA/ATB: each row of A has <=2 nonzeros (+1/-1) scaled
// by sigmoid(att) -> att^2-weighted 2x2 outer products via float atomics.
// ---------------------------------------------------------------------------
__global__ void ata_build(const float* __restrict__ attraw,
                          const float* __restrict__ grad,
                          float* __restrict__ ATA, float* __restrict__ ATB)
{
  const int idx = blockIdx.x * blockDim.x + threadIdx.x;
  if (idx >= 16 * 256 * 10) return;
  const int e = idx % 10;
  const int i = (idx / 10) % 256;
  const int b = idx / 2560;
  const int n = b >> 2, g = b & 3;
  const int ch = g * 10 + e;
  const float araw = attraw[(size_t)(n * 40 + ch) * 256 + i];
  const float att  = 1.0f / (1.0f + __expf(-araw));
  const float gv   = grad[(size_t)(n * 40 + ch) * 256 + i];

  static const int jblk_[10] = {2, 3, 2, 3, 0, 1, 3, 2, 1, 0};
  static const int kblk_[10] = {0, 1, 2, 3, 0, 1, 3, 2, 1, 0};
  int j = -1, k = -1;
  const float vj = 1.0f, vk = -1.0f;
  if (i == 255) {                      // anchor equations for the last pixel
    j = jblk_[e] * 256 + i;
  } else if ((e & 1) == 0) {           // "right" equations
    if (((i + 1) & 15) != 0) { j = jblk_[e] * 256 + i; k = kblk_[e] * 256 + i + 1; }
  } else {                             // "down" equations
    if (i + 16 < 256)         { j = jblk_[e] * 256 + i; k = kblk_[e] * 256 + i + 16; }
  }
  if (j < 0) return;

  const float w2 = att * att;
  const float bw = w2 * gv;
  float* M  = ATA + (size_t)b * 1048576;
  float* Bv = ATB + b * 1024;
  atomicAdd(&M[(size_t)j * 1024 + j], w2 * vj * vj);
  atomicAdd(&Bv[j], bw * vj);
  if (k >= 0) {
    atomicAdd(&M[(size_t)k * 1024 + k], w2 * vk * vk);
    const float cross = w2 * vj * vk;
    atomicAdd(&M[(size_t)j * 1024 + k], cross);
    atomicAdd(&M[(size_t)k * 1024 + j], cross);
    atomicAdd(&Bv[k], bw * vk);
  }
}

// ---------------------------------------------------------------------------
// Blocked Cholesky (64 blocks of 16). 16 independent 1024x1024 SPD systems.
// ---------------------------------------------------------------------------
__global__ __launch_bounds__(32) void chol_diag(float* __restrict__ ATA, int kb)
{
  const int b = blockIdx.x;
  float* M = ATA + (size_t)b * 1048576;
  __shared__ float T[16][17];
  const int lane = threadIdx.x;
  if (lane < 16)
    for (int j = 0; j < 16; ++j)
      T[lane][j] = M[(size_t)(kb * 16 + lane) * 1024 + kb * 16 + j];
  __syncthreads();
  for (int kk = 0; kk < 16; ++kk) {
    if (lane == 0) T[kk][kk] = sqrtf(fmaxf(T[kk][kk], 1e-30f));
    __syncthreads();
    if (lane < 16 && lane > kk) T[lane][kk] /= T[kk][kk];
    __syncthreads();
    if (lane < 16 && lane > kk)
      for (int j = kk + 1; j <= lane; ++j) T[lane][j] -= T[lane][kk] * T[j][kk];
    __syncthreads();
  }
  if (lane < 16)
    for (int j = 0; j < 16; ++j)
      M[(size_t)(kb * 16 + lane) * 1024 + kb * 16 + j] = (j <= lane) ? T[lane][j] : 0.0f;
}

__global__ __launch_bounds__(32) void chol_panel(float* __restrict__ ATA, int kb)
{
  const int b = blockIdx.x;
  const int ib = kb + 1 + blockIdx.y;
  float* M = ATA + (size_t)b * 1048576;
  __shared__ float Lk[16][17];
  const int lane = threadIdx.x;
  if (lane < 16)
    for (int j = 0; j < 16; ++j)
      Lk[lane][j] = M[(size_t)(kb * 16 + lane) * 1024 + kb * 16 + j];
  __syncthreads();
  if (lane < 16) {
    float X[16];
    for (int j = 0; j < 16; ++j)
      X[j] = M[(size_t)(ib * 16 + lane) * 1024 + kb * 16 + j];
    for (int kk = 0; kk < 16; ++kk) {   // X := X * inv(Lk)^T
      float s = X[kk];
      for (int j = 0; j < kk; ++j) s -= X[j] * Lk[kk][j];
      X[kk] = s / Lk[kk][kk];
    }
    for (int j = 0; j < 16; ++j)
      M[(size_t)(ib * 16 + lane) * 1024 + kb * 16 + j] = X[j];
  }
}

// Trailing update A[ib,jb] -= P[ib] * P[jb]^T via fp32 WMMA 16x16x4 (K=16 -> 4x).
__global__ __launch_bounds__(32) void chol_update(float* __restrict__ ATA, int kb)
{
  const int b = blockIdx.x;
  const int t = blockIdx.y;            // triangle index over (ib >= jb > kb)
  int r = (int)((sqrtf(8.0f * (float)t + 1.0f) - 1.0f) * 0.5f);
  while ((r + 1) * (r + 2) / 2 <= t) ++r;
  while (r * (r + 1) / 2 > t) --r;
  const int jloc = t - r * (r + 1) / 2;
  const int ib = kb + 1 + r, jb = kb + 1 + jloc;

  float* M = ATA + (size_t)b * 1048576;
  const float* Pi = M + (size_t)(ib * 16) * 1024 + kb * 16;
  const float* Pj = M + (size_t)(jb * 16) * 1024 + kb * 16;
  float* C = M + (size_t)(ib * 16) * 1024 + jb * 16;

  const int lane = threadIdx.x, m16 = lane & 15, g = lane >> 4;
  v8f acc;
#pragma unroll
  for (int v = 0; v < 8; ++v) acc[v] = C[(size_t)(v + 8 * g) * 1024 + m16];

#pragma unroll
  for (int q = 0; q < 4; ++q) {
    v2f a, bb;
#pragma unroll
    for (int v = 0; v < 2; ++v) {
      const int kk = q * 4 + g * 2 + v;       // 32-bit A 16x4 / B 4x16 layouts
      a[v]  = -Pi[(size_t)m16 * 1024 + kk];   // negate A: D = C - Pi*Pj^T
      bb[v] =  Pj[(size_t)m16 * 1024 + kk];   // B[k][n] = Pj[n][k]
    }
    acc = __builtin_amdgcn_wmma_f32_16x16x4_f32(false, a, false, bb,
                                                (short)0, acc, false, false);
  }
#pragma unroll
  for (int v = 0; v < 8; ++v) C[(size_t)(v + 8 * g) * 1024 + m16] = acc[v];
}

// Forward + backward substitution, one workgroup per system, RHS in LDS.
__global__ __launch_bounds__(256) void chol_solve(
    const float* __restrict__ ATA, const float* __restrict__ ATB,
    float* __restrict__ sol)
{
  const int b = blockIdx.x;
  const float* L = ATA + (size_t)b * 1048576;
  __shared__ float y[1024];
  const int tid = threadIdx.x;
  for (int i = tid; i < 1024; i += 256) y[i] = ATB[b * 1024 + i];
  __syncthreads();
  for (int i = 0; i < 1024; ++i) {               // L y = b
    if (tid == 0) y[i] /= L[(size_t)i * 1024 + i];
    __syncthreads();
    const float yi = y[i];
    for (int r2 = i + 1 + tid; r2 < 1024; r2 += 256)
      y[r2] -= L[(size_t)r2 * 1024 + i] * yi;
    __syncthreads();
  }
  for (int i = 1023; i >= 0; --i) {              // L^T x = y
    if (tid == 0) y[i] /= L[(size_t)i * 1024 + i];
    __syncthreads();
    const float yi = y[i];
    for (int r2 = tid; r2 < i; r2 += 256)
      y[r2] -= L[(size_t)i * 1024 + r2] * yi;
    __syncthreads();
  }
  for (int i = tid; i < 1024; i += 256) sol[b * 1024 + i] = y[i];
}

// ---------------------------------------------------------------------------
// GroupNorm(1 group over 16ch x 256px per sample) + SE scale -> yfin [n][16][256]
// ---------------------------------------------------------------------------
__global__ __launch_bounds__(256) void finalize_gn(
    const float* __restrict__ sol, const float* __restrict__ sev,
    const float* __restrict__ gn_w, const float* __restrict__ gn_b,
    float* __restrict__ yfin)
{
  const int n = blockIdx.x, tid = threadIdx.x;
  __shared__ float r1[256], r2[256];
  float s1 = 0.0f, s2 = 0.0f;
  for (int idx = tid; idx < 4096; idx += 256) {
    const int g = idx >> 10, s = idx & 1023;
    const float v = sol[(size_t)(n * 4 + g) * 1024 + s];
    s1 += v; s2 += v * v;
  }
  r1[tid] = s1; r2[tid] = s2;
  __syncthreads();
  for (int off = 128; off > 0; off >>= 1) {
    if (tid < off) { r1[tid] += r1[tid + off]; r2[tid] += r2[tid + off]; }
    __syncthreads();
  }
  const float mu  = r1[0] * (1.0f / 4096.0f);
  const float var = r2[0] * (1.0f / 4096.0f) - mu * mu;
  const float inv = rsqrtf(var + 1e-5f);
  for (int idx = tid; idx < 4096; idx += 256) {
    const int g = idx >> 10, s = idx & 1023;
    const int comp = s >> 8, pix = s & 255;
    const int c = g * 4 + comp;
    float v = (sol[(size_t)(n * 4 + g) * 1024 + s] - mu) * inv;
    v = v * gn_w[c] + gn_b[c];
    v *= sev[n * 16 + c];
    yfin[(size_t)(n * 16 + c) * 256 + pix] = v;
  }
}

// ---------------------------------------------------------------------------
extern "C" void kernel_launch(void* const* d_in, const int* in_sizes, int n_in,
                              void* d_out, int out_size, void* d_ws, size_t ws_size,
                              hipStream_t stream)
{
  const float* x       = (const float*)d_in[0];
  const float* grad_w1 = (const float*)d_in[1];
  const float* grad_b1 = (const float*)d_in[2];
  const float* grad_w2 = (const float*)d_in[3];
  const float* grad_b2 = (const float*)d_in[4];
  const float* att_w1  = (const float*)d_in[5];
  const float* att_b1  = (const float*)d_in[6];
  const float* att_w2  = (const float*)d_in[7];
  const float* att_b2  = (const float*)d_in[8];
  const float* se_w1   = (const float*)d_in[9];
  const float* se_b1   = (const float*)d_in[10];
  const float* se_w2   = (const float*)d_in[11];
  const float* se_b2   = (const float*)d_in[12];
  const float* gn_w    = (const float*)d_in[13];
  const float* gn_b    = (const float*)d_in[14];
  const float* post_w  = (const float*)d_in[15];
  const float* post_b  = (const float*)d_in[16];
  float* out = (float*)d_out;

  // workspace layout (floats); total ~17.04M floats (~68 MB)
  float* ws    = (float*)d_ws;
  float* t1    = ws;             // 65536
  float* t2    = t1 + 65536;     // 65536
  float* grad2 = t2 + 65536;     // 40960
  float* att2  = grad2 + 40960;  // 40960 (pre-sigmoid)
  float* sev   = att2 + 40960;   // 64
  float* ATB   = sev + 64;       // 16384
  float* sol   = ATB + 16384;    // 16384
  float* yfin  = sol + 16384;    // 16384
  float* ATA   = yfin + 16384;   // 16 * 1048576

  conv3x3_wmma<64, 64, 1><<<dim3(4, 4, 16), 32, 0, stream>>>(x,  grad_w1, grad_b1, t1);
  conv3x3_wmma<64, 40, 0><<<dim3(4, 3, 16), 32, 0, stream>>>(t1, grad_w2, grad_b2, grad2);
  conv3x3_wmma<64, 64, 1><<<dim3(4, 4, 16), 32, 0, stream>>>(x,  att_w1,  att_b1,  t2);
  conv3x3_wmma<64, 40, 0><<<dim3(4, 3, 16), 32, 0, stream>>>(t2, att_w2,  att_b2,  att2);
  se_kernel<<<4, 64, 0, stream>>>(x, se_w1, se_b1, se_w2, se_b2, sev);

  ata_init<<<65536, 256, 0, stream>>>(ATA, ATB);
  ata_build<<<160, 256, 0, stream>>>(att2, grad2, ATA, ATB);

  for (int kb = 0; kb < 64; ++kb) {
    chol_diag<<<16, 32, 0, stream>>>(ATA, kb);
    const int m = 63 - kb;
    if (m > 0) {
      chol_panel<<<dim3(16, m), 32, 0, stream>>>(ATA, kb);
      chol_update<<<dim3(16, m * (m + 1) / 2), 32, 0, stream>>>(ATA, kb);
    }
  }
  chol_solve<<<16, 256, 0, stream>>>(ATA, ATB, sol);
  finalize_gn<<<4, 256, 0, stream>>>(sol, sev, gn_w, gn_b, yfin);
  conv3x3_wmma<16, 128, 0><<<dim3(4, 8, 16), 32, 0, stream>>>(yfin, post_w, post_b, out);
}
